// AttentionLayer_14276471292820
// MI455X (gfx1250) — compile-verified, matrix operand order checked by reading, not run on hardware
//
#include <hip/hip_runtime.h>
#include <stdint.h>

// ---------------------------------------------------------------------------
// CDNA5 (gfx1250) wave32 types
// ---------------------------------------------------------------------------
typedef __attribute__((ext_vector_type(16))) _Float16 v16h;
typedef __attribute__((ext_vector_type(8)))  _Float16 v8h;
typedef __attribute__((ext_vector_type(8)))  float    v8f;
typedef __attribute__((ext_vector_type(4)))  float    f32x4;
typedef __attribute__((ext_vector_type(4)))  unsigned int u32x4;
typedef __attribute__((ext_vector_type(8)))  int      i32x8;
typedef __attribute__((ext_vector_type(4)))  int      i32x4;

#if defined(__has_builtin)
#  if __has_builtin(__builtin_amdgcn_tensor_load_to_lds)
#    define HAVE_TDM 1
#  endif
#endif
#ifndef HAVE_TDM
#  define HAVE_TDM 0
#endif

// Fragment (16 rows x 32 K, f16) from a row-major LDS tile, K-contiguous.
// ISA 7.12.2: lane l<16: row=l, K={0..7,16..23}; lane l+16: K={8..15,24..31}.
// Rows are 80B (16B-aligned) => two ds_load_b128 per fragment.
__device__ __forceinline__ v16h frag_row(const _Float16* p, int ld, int lane) {
  const _Float16* q = p + (lane & 15) * ld + ((lane >> 4) << 3);
  v16h f;
#pragma unroll
  for (int j = 0; j < 8; ++j) f[j] = q[j];
#pragma unroll
  for (int j = 0; j < 8; ++j) f[j + 8] = q[16 + j];
  return f;
}

__device__ __forceinline__ v8f wmma_f16(v16h a, v16h b, v8f c) {
  return __builtin_amdgcn_wmma_f32_16x16x32_f16(false, a, false, b, (short)0, c,
                                                false, false);
}

__device__ __forceinline__ unsigned lds_off(const void* p) {
  return (unsigned)(uintptr_t)p;   // LDS aperture: addr[31:0] == LDS byte offset
}

#if HAVE_TDM
// TDM: load a tile_w x tile_h f16 tile from a row-major tensor (row stride
// `stride` elems) into LDS at lds_off, padding (padamt+1) DWORDs after every
// (2<<padiv) DWORDs.  D# layout per CDNA5 ISA ch.8 (group0/group1).
__device__ __forceinline__ void tdm_load_2d(const _Float16* gsrc, unsigned ldsoff,
                                            int tile_w, int tile_h, int stride,
                                            unsigned padiv, unsigned padamt) {
  const unsigned long long ga = (unsigned long long)(uintptr_t)gsrc;
  u32x4 g0;
  g0.x = 1u;                                                 // count = 1 (valid)
  g0.y = ldsoff;                                             // lds_addr
  g0.z = (unsigned)(ga & 0xffffffffull);                     // global_addr lo
  g0.w = (unsigned)((ga >> 32) & 0x1ffffffull) | (2u << 30); // addr[56:32]|type=2
  const unsigned dim = 1u << 20;                             // large logical dims
  i32x8 g1;
  g1[0] = (int)((1u << 16) | (1u << 20) | (padiv << 22) | (padamt << 25));
  g1[1] = (int)((dim & 0xffffu) << 16);                      // tensor_dim0 lo16
  g1[2] = (int)(((dim >> 16) & 0xffffu) | ((dim & 0xffffu) << 16));
  g1[3] = (int)(((dim >> 16) & 0xffffu) | ((unsigned)tile_w << 16));
  g1[4] = tile_h;                                            // tile_dim1 (dim2=0)
  g1[5] = stride;                                            // dim0 stride lo32
  g1[6] = 0;
  g1[7] = 0;
  const i32x4 z4 = {0, 0, 0, 0};
#if __clang_major__ >= 23
  const i32x8 z8 = {0, 0, 0, 0, 0, 0, 0, 0};
  __builtin_amdgcn_tensor_load_to_lds(g0, g1, z4, z4, z8, 0);
#else
  __builtin_amdgcn_tensor_load_to_lds(g0, g1, z4, z4, 0);
#endif
}
#endif

// ---------------------------------------------------------------------------
// Tiled WMMA GEMM:  C[b] = alpha * A[b] x B[b] (+ bias[m]);  N fixed at
// compile time (NFIX) so B staging folds into immediate-offset loads.
//   A: f16 [>=grid.y*128, K] row-major zero-padded; strideA=0 => shared.
//      Staged by the Tensor Data Mover, DOUBLE-BUFFERED in LDS (2x[128][40]).
//   B: !B_TRANS: f16 [K,N] | B_TRANS: f16 [N,K].  Staged TRANSPOSED in LDS
//      ([64][40]) so all fragment reads are K-contiguous ds_load_b128.
//   C: f32 [M,N]; K mult 32; only gm<M rows stored.
// 256 thr = 8 waves (4x2); block tile 128x64; wave tile 32x32; BK=32.
// ---------------------------------------------------------------------------
template <bool B_TRANS, bool HAS_BIAS, int NFIX>
__global__ __launch_bounds__(256) void gemm_wmma_kernel(
    const _Float16* __restrict__ A, const _Float16* __restrict__ Bm,
    const float* __restrict__ bias, float* __restrict__ C,
    int M, int K, long long sA, long long sB, long long sC, float alpha)
{
  constexpr int N = NFIX;
  constexpr int BM = 128, BN = 64, BK = 32, PAD = 8;
  constexpr int LDA = BK + PAD;                    // 40 halfs = 80B rows
  constexpr int ASZ = BM * LDA;
  __shared__ __align__(16) _Float16 As[2 * ASZ];   // double-buffered (TDM)
  __shared__ __align__(16) _Float16 Bs[BN * LDA];

  const int bz = blockIdx.z;
  const long long aoff = sA * bz, boff = sB * bz, coff = sC * bz;
  const int tid = threadIdx.x, lane = tid & 31, wave = tid >> 5;
  const int wm0 = (wave & 3) * 32, wn0 = (wave >> 2) * 32;
  const int m0 = blockIdx.y * BM, n0 = blockIdx.x * BN;

  // B staging coordinates (n within tile, k chunk)
  const int bn = B_TRANS ? (tid >> 2) : (tid & 63);
  const int bk = B_TRANS ? ((tid & 3) * 8) : ((tid >> 6) * 8);

  const _Float16* aptr = A + aoff + (long long)m0 * K;
  const _Float16* bptr = B_TRANS ? (Bm + boff + (long long)(n0 + bn) * K + bk)
                                 : (Bm + boff + (long long)bk * N + n0 + bn);
  const long long bstep = B_TRANS ? (long long)BK : (long long)BK * N;
#if !HAVE_TDM
  const int ar = tid >> 2, ac = (tid & 3) * 8;
#endif

  v8f acc[2][2] = {};

#if HAVE_TDM
  if (wave == 0)  // prologue: DMA first A tile into buffer 0
    tdm_load_2d(aptr, lds_off(&As[0]), BK, BM, K, 3u, 3u);
#endif

  for (int k0 = 0; k0 < K; k0 += BK) {
    _Float16* Asc = &As[((k0 >> 5) & 1) * ASZ];
#if HAVE_TDM
    if (wave == 0) {
      if (k0 + BK < K) {  // DMA next tile into the other buffer, keep 1 in flight
        tdm_load_2d(aptr + (k0 + BK), lds_off(&As[(((k0 >> 5) + 1) & 1) * ASZ]),
                    BK, BM, K, 3u, 3u);
        __builtin_amdgcn_s_wait_tensorcnt(1);
      } else {
        __builtin_amdgcn_s_wait_tensorcnt(0);
      }
    }
#else
    *(v8h*)&Asc[ar * LDA + ac] = *(const v8h*)&aptr[(long long)ar * K + k0 + ac];
    *(v8h*)&Asc[(ar + 64) * LDA + ac] =
        *(const v8h*)&aptr[(long long)(ar + 64) * K + k0 + ac];
#endif
    // ---- B tile -> transposed LDS [BN][LDA] ----
    if (B_TRANS) {
      *(v8h*)&Bs[bn * LDA + bk] = *(const v8h*)bptr;
    } else {
      v8h vv;  // 8 lane-coalesced u16 loads (immediate offsets j*N), one b128 store
#pragma unroll
      for (int j = 0; j < 8; ++j) vv[j] = bptr[(long long)j * N];
      *(v8h*)&Bs[bn * LDA + bk] = vv;
    }
    if (k0 + BK < K)  // L2 prefetch of next B tile (global_prefetch_b8)
      __builtin_prefetch((const void*)(bptr + bstep), 0, 1);
    bptr += bstep;
    __syncthreads();

    const v16h a0 = frag_row(&Asc[wm0 * LDA],        LDA, lane);
    const v16h a1 = frag_row(&Asc[(wm0 + 16) * LDA], LDA, lane);
    const v16h b0 = frag_row(&Bs[wn0 * LDA],         LDA, lane);
    const v16h b1 = frag_row(&Bs[(wn0 + 16) * LDA],  LDA, lane);
    acc[0][0] = wmma_f16(a0, b0, acc[0][0]);
    acc[0][1] = wmma_f16(a0, b1, acc[0][1]);
    acc[1][0] = wmma_f16(a1, b0, acc[1][0]);
    acc[1][1] = wmma_f16(a1, b1, acc[1][1]);
    __syncthreads();
  }

  // ---- epilogue: D layout: vgpr r -> M=r (+8 for lanes>=16), N = lane&15 ----
  const int mhalf = (lane >= 16) ? 8 : 0;
  const int nn    = lane & 15;
  if (m0 + BM <= M) {  // fast path: whole tile in range, no per-store guards
#pragma unroll
    for (int ti = 0; ti < 2; ++ti)
#pragma unroll
      for (int tj = 0; tj < 2; ++tj) {
        const int gn = n0 + wn0 + tj * 16 + nn;
#pragma unroll
        for (int r = 0; r < 8; ++r) {
          const int gm = m0 + wm0 + ti * 16 + mhalf + r;
          float v = acc[ti][tj][r] * alpha;
          if (HAS_BIAS) v += bias[gm];
          C[coff + (long long)gm * N + gn] = v;
        }
      }
  } else {
#pragma unroll
    for (int ti = 0; ti < 2; ++ti)
#pragma unroll
      for (int tj = 0; tj < 2; ++tj) {
        const int gn = n0 + wn0 + tj * 16 + nn;
#pragma unroll
        for (int r = 0; r < 8; ++r) {
          const int gm = m0 + wm0 + ti * 16 + mhalf + r;
          if (gm < M) {
            float v = acc[ti][tj][r] * alpha;
            if (HAS_BIAS) v += bias[gm];
            C[coff + (long long)gm * N + gn] = v;
          }
        }
      }
  }
}

// ---------------------------------------------------------------------------
// Channel LayerNorm (+optional exact GELU). f32 [Sin,L] in -> f16/f32 [Sout,L].
// ---------------------------------------------------------------------------
template <bool GELU, bool OUT_F16>
__global__ __launch_bounds__(256) void ln_chan_kernel(
    const float* __restrict__ x, const float* __restrict__ g,
    const float* __restrict__ be, void* __restrict__ out,
    int C, int Sin, int Sout, int L)
{
  const long long b = blockIdx.z;
  const int p = blockIdx.x * 256 + threadIdx.x;
  const float* xb = x + b * (long long)Sin * L + p;
  float s = 0.f, ss = 0.f;
  for (int c = 0; c < C; ++c) { float v = xb[(long long)c * L]; s += v; ss += v * v; }
  const float mean = s / (float)C;
  float var = ss / (float)C - mean * mean;
  var = var < 0.f ? 0.f : var;
  const float rstd = rsqrtf(var + 1e-6f);
  _Float16* oh = (_Float16*)out;
  float*    of = (float*)out;
  for (int c = 0; c < C; ++c) {
    float v = (xb[(long long)c * L] - mean) * rstd * g[c] + be[c];
    if (GELU) v = 0.5f * v * (1.f + erff(v * 0.70710678118654752f));
    const long long o = b * (long long)Sout * L + (long long)c * L + p;
    if (OUT_F16) oh[o] = (_Float16)v; else of[o] = v;
  }
}

// 3x3 depthwise conv, SAME padding; f16 in [Cs,L], f32 out [Cs,L].
__global__ __launch_bounds__(256) void dwconv3x3_kernel(
    const _Float16* __restrict__ x, const float* __restrict__ w,
    const float* __restrict__ bias, float* __restrict__ y,
    int Cs, int H, int W)
{
  const int b = blockIdx.z, c = blockIdx.y;
  const int p = blockIdx.x * 256 + threadIdx.x;
  const int h = p >> 6, wp = p & 63;
  const _Float16* xc = x + ((long long)b * Cs + c) * (long long)(H * W);
  const float* wc = w + c * 9;
  float acc = bias[c];
#pragma unroll
  for (int dy = -1; dy <= 1; ++dy)
#pragma unroll
    for (int dx = -1; dx <= 1; ++dx) {
      const int hh = h + dy, ww = wp + dx;
      if (hh >= 0 && hh < H && ww >= 0 && ww < W)
        acc += (float)xc[hh * W + ww] * wc[(dy + 1) * 3 + (dx + 1)];
    }
  y[((long long)b * Cs + c) * (long long)(H * W) + p] = acc;
}

// f32 [C(Ss),L] -> f16 [C(Ds),L] with channel padding (pad rows pre-zeroed).
__global__ __launch_bounds__(256) void pack_pad_kernel(
    const float* __restrict__ src, _Float16* __restrict__ dst,
    int Ss, int Ds, int L)
{
  const long long b = blockIdx.z; const int c = blockIdx.y;
  const int p = blockIdx.x * 256 + threadIdx.x;
  dst[(b * Ds + c) * (long long)L + p] = (_Float16)src[(b * Ss + c) * (long long)L + p];
}

// f32 [M,K] weights -> f16 [Mp,Kp] zero-padded.
__global__ __launch_bounds__(256) void pack_weight_kernel(
    const float* __restrict__ src, _Float16* __restrict__ dst,
    int M, int K, int Kp)
{
  const int k = blockIdx.x * 256 + threadIdx.x;
  const int m = blockIdx.y;
  if (k >= Kp) return;
  const float v = (m < M && k < K) ? src[(long long)m * K + k] : 0.f;
  dst[(long long)m * Kp + k] = (_Float16)v;
}

__global__ void fill_zero_kernel(f32x4* __restrict__ p, long long n4) {
  long long i = (long long)blockIdx.x * blockDim.x + threadIdx.x;
  const long long stride = (long long)gridDim.x * blockDim.x;
  const f32x4 z = {0.f, 0.f, 0.f, 0.f};
  for (; i < n4; i += stride) p[i] = z;
}

// softmax over cols for rows (b, q<QC), row stride QP.
__global__ __launch_bounds__(256) void softmax_rows_kernel(
    const float* __restrict__ x, _Float16* __restrict__ out,
    int QC, int QP, int cols)
{
  const int row = blockIdx.x;
  const int b = row / QC, q = row - b * QC;
  const long long base = ((long long)b * QP + q) * cols;
  const float* xr = x + base;
  __shared__ float red[256];
  const int tid = threadIdx.x;
  float mx = -3.4e38f;
  for (int c = tid; c < cols; c += 256) mx = fmaxf(mx, xr[c]);
  red[tid] = mx; __syncthreads();
  for (int s = 128; s > 0; s >>= 1) {
    if (tid < s) red[tid] = fmaxf(red[tid], red[tid + s]);
    __syncthreads();
  }
  const float rowmax = red[0]; __syncthreads();
  float sum = 0.f;
  for (int c = tid; c < cols; c += 256) sum += __expf(xr[c] - rowmax);
  red[tid] = sum; __syncthreads();
  for (int s = 128; s > 0; s >>= 1) {
    if (tid < s) red[tid] += red[tid + s];
    __syncthreads();
  }
  const float inv = 1.f / red[0];
  for (int c = tid; c < cols; c += 256)
    out[base + c] = (_Float16)(__expf(xr[c] - rowmax) * inv);
}

__global__ void residual_kernel(const float* __restrict__ a,
                                const float* __restrict__ b,
                                float* __restrict__ o, long long n) {
  const long long i = (long long)blockIdx.x * blockDim.x + threadIdx.x;
  if (i < n) o[i] = a[i] + b[i];
}

// ---------------------------------------------------------------------------
// Host orchestration
// ---------------------------------------------------------------------------
namespace {
constexpr int B_ = 8, QC = 150, QP = 160, QW = 256, KC = 768, H_ = 64, W_ = 64;
constexpr int L_ = H_ * W_;

struct DevBlock {
  const _Float16 *w1p, *w3p;                      // packed f16 weights
  const float *b1, *g1, *be1, *w2, *b2, *g2, *be2, *b3, *g3, *be3;
};

void run_dw_block(const DevBlock& P, const _Float16* x_h, int C, int Cp,
                  float* t_f32, _Float16* th0, _Float16* th1,
                  _Float16* out_h, float* out_f, int SoutF, hipStream_t stream) {
  const dim3 blk(256);
  const long long sAct = (long long)Cp * L_;
  const dim3 gg(L_ / 64, (C + 127) / 128, B_);
  gemm_wmma_kernel<false, true, L_><<<gg, blk, 0, stream>>>(
      P.w1p, x_h, P.b1, t_f32, C, Cp, 0LL, sAct, sAct, 1.0f);
  const dim3 gl(L_ / 256, 1, B_);
  ln_chan_kernel<true, true><<<gl, blk, 0, stream>>>(t_f32, P.g1, P.be1, th0, C, Cp, Cp, L_);
  const dim3 gd(L_ / 256, C, B_);
  dwconv3x3_kernel<<<gd, blk, 0, stream>>>(th0, P.w2, P.b2, t_f32, Cp, H_, W_);
  ln_chan_kernel<true, true><<<gl, blk, 0, stream>>>(t_f32, P.g2, P.be2, th1, C, Cp, Cp, L_);
  gemm_wmma_kernel<false, true, L_><<<gg, blk, 0, stream>>>(
      P.w3p, th1, P.b3, t_f32, C, Cp, 0LL, sAct, sAct, 1.0f);
  if (out_f)
    ln_chan_kernel<false, false><<<gl, blk, 0, stream>>>(t_f32, P.g3, P.be3, out_f, C, Cp, SoutF, L_);
  else
    ln_chan_kernel<false, true><<<gl, blk, 0, stream>>>(t_f32, P.g3, P.be3, out_h, C, Cp, Cp, L_);
}
} // namespace

extern "C" void kernel_launch(void* const* d_in, const int* in_sizes, int n_in,
                              void* d_out, int out_size, void* d_ws, size_t ws_size,
                              hipStream_t stream) {
  (void)in_sizes; (void)n_in; (void)out_size; (void)ws_size;
  const float* query = (const float*)d_in[0];
  const float* key   = (const float*)d_in[1];
  const float* value = (const float*)d_in[2];
  auto F = [&](int i) { return (const float*)d_in[i]; };

  char* ws = (char*)d_ws;
  auto carve = [&](size_t bytes) -> char* {
    char* p = ws; ws += (bytes + 255) & ~(size_t)255; return p;
  };
  float*    t_f32  = (float*)   carve((size_t)B_ * KC * L_ * 4);
  _Float16* th0    = (_Float16*)carve((size_t)B_ * KC * L_ * 2);
  _Float16* th1    = (_Float16*)carve((size_t)B_ * KC * L_ * 2);
  _Float16* q_h    = (_Float16*)carve((size_t)B_ * QP * L_ * 2);
  _Float16* k_h    = (_Float16*)carve((size_t)B_ * KC * L_ * 2);
  _Float16* v_h    = (_Float16*)carve((size_t)B_ * KC * L_ * 2);
  float*    attn_f = (float*)   carve((size_t)B_ * QP * KC * 4);
  _Float16* attn_h = (_Float16*)carve((size_t)B_ * QP * KC * 2);
  float*    x_f    = (float*)   carve((size_t)B_ * QC * L_ * 4);
  _Float16* wq1 = (_Float16*)carve((size_t)QW * QP * 2);
  _Float16* wq3 = (_Float16*)carve((size_t)QW * QP * 2);
  _Float16* wk1 = (_Float16*)carve((size_t)KC * KC * 2);
  _Float16* wk3 = (_Float16*)carve((size_t)KC * KC * 2);
  _Float16* wv1 = (_Float16*)carve((size_t)KC * KC * 2);
  _Float16* wv3 = (_Float16*)carve((size_t)KC * KC * 2);
  _Float16* wo1 = (_Float16*)carve((size_t)QW * QP * 2);
  _Float16* wo3 = (_Float16*)carve((size_t)QW * QP * 2);
  const long long total16 = (long long)(ws - (char*)d_ws) / 16;

  // 1) zero the workspace (pad rows must be finite/zero; deterministic per call)
  fill_zero_kernel<<<4096, 256, 0, stream>>>((f32x4*)d_ws, total16);

  // 2) pack all 1x1-conv weights to zero-padded f16
  pack_weight_kernel<<<dim3(1, QW), 256, 0, stream>>>(F(3),  wq1, QC, QC, QP);
  pack_weight_kernel<<<dim3(1, QW), 256, 0, stream>>>(F(11), wq3, QC, QC, QP);
  pack_weight_kernel<<<dim3(3, KC), 256, 0, stream>>>(F(15), wk1, KC, KC, KC);
  pack_weight_kernel<<<dim3(3, KC), 256, 0, stream>>>(F(23), wk3, KC, KC, KC);
  pack_weight_kernel<<<dim3(3, KC), 256, 0, stream>>>(F(27), wv1, KC, KC, KC);
  pack_weight_kernel<<<dim3(3, KC), 256, 0, stream>>>(F(35), wv3, KC, KC, KC);
  pack_weight_kernel<<<dim3(1, QW), 256, 0, stream>>>(F(39), wo1, QC, QC, QP);
  pack_weight_kernel<<<dim3(1, QW), 256, 0, stream>>>(F(47), wo3, QC, QC, QP);

  const DevBlock qp = {wq1, wq3, F(4),  F(5),  F(6),  F(7),  F(8),  F(9),  F(10), F(12), F(13), F(14)};
  const DevBlock kp = {wk1, wk3, F(16), F(17), F(18), F(19), F(20), F(21), F(22), F(24), F(25), F(26)};
  const DevBlock vp = {wv1, wv3, F(28), F(29), F(30), F(31), F(32), F(33), F(34), F(36), F(37), F(38)};
  const DevBlock op = {wo1, wo3, F(40), F(41), F(42), F(43), F(44), F(45), F(46), F(48), F(49), F(50)};

  // 3) q / k / v blocks
  pack_pad_kernel<<<dim3(L_ / 256, QC, B_), 256, 0, stream>>>(query, th0, QC, QP, L_);
  run_dw_block(qp, th0, QC, QP, t_f32, th0, th1, q_h, nullptr, 0, stream);
  pack_pad_kernel<<<dim3(L_ / 256, KC, B_), 256, 0, stream>>>(key, th0, KC, KC, L_);
  run_dw_block(kp, th0, KC, KC, t_f32, th0, th1, k_h, nullptr, 0, stream);
  pack_pad_kernel<<<dim3(L_ / 256, KC, B_), 256, 0, stream>>>(value, th0, KC, KC, L_);
  run_dw_block(vp, th0, KC, KC, t_f32, th0, th1, v_h, nullptr, 0, stream);

  // 4) attn = (q @ k^T) / sqrt(L)  : M=QC, N=KC, K=L  (B = k, [N,K] layout)
  gemm_wmma_kernel<true, false, KC><<<dim3(KC / 64, (QC + 127) / 128, B_), 256, 0, stream>>>(
      q_h, k_h, nullptr, attn_f, QC, L_,
      (long long)QP * L_, (long long)KC * L_, (long long)QP * KC, 0.015625f);
  softmax_rows_kernel<<<B_ * QC, 256, 0, stream>>>(attn_f, attn_h, QC, QP, KC);

  // 5) x = attn @ v : M=QC, N=L, K=KC
  gemm_wmma_kernel<false, false, L_><<<dim3(L_ / 64, (QC + 127) / 128, B_), 256, 0, stream>>>(
      attn_h, v_h, nullptr, t_f32, QC, KC,
      (long long)QP * KC, (long long)KC * L_, (long long)QP * L_, 1.0f);

  // 6) output block + residual
  pack_pad_kernel<<<dim3(L_ / 256, QC, B_), 256, 0, stream>>>(t_f32, th0, QP, QP, L_);
  run_dw_block(op, th0, QC, QP, t_f32, th0, th1, nullptr, x_f, QC, stream);
  const long long nq = (long long)B_ * QC * L_;
  residual_kernel<<<(unsigned)((nq + 255) / 256), 256, 0, stream>>>(
      query, x_f, (float*)d_out, nq);
}
